// MagnetAttention_40372692583003
// MI455X (gfx1250) — matrix operand order repaired and checked by reading, not compiled, not run on hardware
//
#include <hip/hip_runtime.h>
#include <hip/hip_bf16.h>

typedef __attribute__((ext_vector_type(16))) _Float16 v16h;
typedef __attribute__((ext_vector_type(8)))  _Float16 v8h;
typedef __attribute__((ext_vector_type(8)))  float    v8f;
typedef __attribute__((ext_vector_type(4)))  unsigned int v4u;
typedef __attribute__((ext_vector_type(8)))  int v8i;
typedef __attribute__((ext_vector_type(4)))  int v4i;

#define NB    4
#define TSEQ  2048
#define DEMB  1024
#define NH    16
#define HD    64
#define NTOK  (NB * TSEQ)      /* 8192 */
#define QK_SCALE 0.125f        /* 64^-0.5 */

// ---------------------------------------------------------------------------
// WMMA helper (CDNA5 wave32, v_wmma_f32_16x16x32_f16)
// ---------------------------------------------------------------------------
__device__ __forceinline__ v8f wmma_f16(v16h a, v16h b, v8f c) {
  return __builtin_amdgcn_wmma_f32_16x16x32_f16(false, a, false, b, (short)0, c,
                                                false, false);
}

// A fragment: 16x32 f16 tile, row-major source with leading dim `ld`.
// lanes 0-15 -> M=lane, K in {0..7,16..23}; lanes 16-31 -> M=lane-16,
// K in {8..15,24..31}. Two 128-bit loads per lane (global or LDS).
__device__ __forceinline__ v16h load_a_frag(const _Float16* src, int ld) {
  int lane = threadIdx.x & 31;
  int r    = lane & 15;
  int kb   = (lane >> 4) << 3;               // 0 or 8
  const _Float16* p = src + (size_t)r * ld + kb;
  v8h lo = *(const v8h*)p;                   // K = kb .. kb+7
  v8h hi = *(const v8h*)(p + 16);            // K = 16+kb .. 16+kb+7
  v16h a;
#pragma unroll
  for (int i = 0; i < 8; ++i) { a[i] = lo[i]; a[8 + i] = hi[i]; }
  return a;
}

// B fragment from a K-contiguous operand bt[N][K]:
// N = lane&15, lanes 0-15 -> K in {0..7,16..23}; lanes 16-31 -> K in {8..15,24..31}. Two 128-bit loads.
__device__ __forceinline__ v16h load_bt_frag(const _Float16* bt, int ld) {
  int lane = threadIdx.x & 31;
  int n    = lane & 15;
  int kb   = (lane >> 4) << 3;               // 0 or 8
  const _Float16* p = bt + (size_t)n * ld + kb;
  v8h lo = *(const v8h*)p;
  v8h hi = *(const v8h*)(p + 16);
  v16h b;
#pragma unroll
  for (int i = 0; i < 8; ++i) { b[i] = lo[i]; b[8 + i] = hi[i]; }
  return b;
}

// ---------------------------------------------------------------------------
// Tensor Data Mover: async 2-D tile load Global -> LDS (D# per ISA 8.3/8.4).
// tile_d0 = contiguous elements per row, tile_d1 = rows; f16 (data_size=1).
// Tracked by TENSORcnt.  (6-arg builtin form: amdgpu-toolchain / clang-23)
// ---------------------------------------------------------------------------
__device__ __forceinline__ unsigned lds_off(const void* p) {
  return (unsigned)(uintptr_t)p;             // low 32 bits = LDS offset
}

__device__ __forceinline__ void tdm_load_2d(const _Float16* gptr, unsigned ldsaddr,
                                            unsigned tile_d0, unsigned tile_d1,
                                            unsigned tensor_d0, unsigned tensor_d1,
                                            unsigned stride0) {
  unsigned long long ga = (unsigned long long)(uintptr_t)gptr;
  v4u g0;
  g0[0] = 1u;                                            // count=1, user D#
  g0[1] = ldsaddr;                                       // lds_addr (bytes)
  g0[2] = (unsigned)(ga & 0xFFFFFFFFu);                  // global_addr[31:0]
  g0[3] = (unsigned)((ga >> 32) & 0x01FFFFFFu) | (2u << 30);  // addr[56:32]|type=2
  v8i g1;
  g1[0] = (int)(1u << 16);                               // data_size=1 (2B), mask=0
  g1[1] = (int)((tensor_d0 & 0xFFFFu) << 16);            // tensor_dim0[15:0]
  g1[2] = (int)((tensor_d0 >> 16) | ((tensor_d1 & 0xFFFFu) << 16));
  g1[3] = (int)((tensor_d1 >> 16) | (tile_d0 << 16));    // tensor_dim1 hi | tile_dim0
  g1[4] = (int)(tile_d1 & 0xFFFFu);                      // tile_dim1 (tile_dim2=0)
  g1[5] = (int)stride0;                                  // tensor_dim0_stride[31:0]
  g1[6] = 0;                                             // stride0 hi | stride1 lo
  g1[7] = 0;
  v4i z4 = {0, 0, 0, 0};
  v8i z8 = {0, 0, 0, 0, 0, 0, 0, 0};
  __builtin_amdgcn_tensor_load_to_lds(g0, g1, z4, z4, z8, 0);
}

// ---------------------------------------------------------------------------
// Conversion kernels
// ---------------------------------------------------------------------------
__global__ void cvt_f32_to_f16(const float* __restrict__ src,
                               _Float16* __restrict__ dst, int n) {
  int i = blockIdx.x * blockDim.x + threadIdx.x;
  if (i < n) dst[i] = (_Float16)src[i];
}

// w: [Din][Dout] row-major -> wt: [Dout][Din] f16 (K-contiguous B operand)
__global__ void cvt_transpose_w(const float* __restrict__ w,
                                _Float16* __restrict__ wt) {
  int k = blockIdx.x * blockDim.x + threadIdx.x;   // Din
  int n = blockIdx.y * blockDim.y + threadIdx.y;   // Dout
  wt[(size_t)n * DEMB + k] = (_Float16)w[(size_t)k * DEMB + n];
}

// ---------------------------------------------------------------------------
// Projection GEMM: TDM-staged, double-buffered 64x64 LDS tiles, K-stage = 64.
// Block = 4 waves -> 64x64 output tile; wave w owns rows w*16..w*16+15.
// Inner loop software-pipelined: B-frags for tile n+1 issued under WMMAs of n.
// mode 0: f16 q/k layout [B*H][T][Hd]; mode 1: f16 v transposed [B*H][Hd][T];
// mode 2: fp32 flat [NTOK][DEMB].
// ---------------------------------------------------------------------------
__global__ void __launch_bounds__(128)
proj_gemm(const _Float16* __restrict__ X, const _Float16* __restrict__ Wt,
          const float* __restrict__ bias, _Float16* __restrict__ out_h,
          float* __restrict__ out_f, float scale, int mode) {
  __shared__ _Float16 sX[2][64 * 64];    // [row][k] tiles, double buffered
  __shared__ _Float16 sW[2][64 * 64];    // [col][k] tiles

  int wave = threadIdx.x >> 5;
  int lane = threadIdx.x & 31;
  int row0 = blockIdx.x * 64;
  int col0 = blockIdx.y * 64;
  const int NKT = DEMB / 64;             // 16 K-stages

  if (wave == 0) {                       // stage 0
    tdm_load_2d(X + (size_t)row0 * DEMB, lds_off(&sX[0][0]), 64, 64, DEMB, NTOK, DEMB);
    tdm_load_2d(Wt + (size_t)col0 * DEMB, lds_off(&sW[0][0]), 64, 64, DEMB, DEMB, DEMB);
  }

  v8f acc[4] = {};
  for (int i = 0; i < NKT; ++i) {
    int cur = i & 1, nxt = cur ^ 1;
    if (wave == 0) {
      if (i + 1 < NKT) {
        int kt = (i + 1) * 64;
        tdm_load_2d(X + (size_t)row0 * DEMB + kt, lds_off(&sX[nxt][0]), 64, 64,
                    DEMB, NTOK, DEMB);
        tdm_load_2d(Wt + (size_t)col0 * DEMB + kt, lds_off(&sW[nxt][0]), 64, 64,
                    DEMB, DEMB, DEMB);
        __builtin_amdgcn_s_wait_tensorcnt(2);   // stage i complete
      } else {
        __builtin_amdgcn_s_wait_tensorcnt(0);
      }
    }
    __syncthreads();

    const _Float16* xs = &sX[cur][wave * 16 * 64];
    const _Float16* ws = &sW[cur][0];
    v16h a0 = load_a_frag(xs, 64);               // K 0..31
    v16h a1 = load_a_frag(xs + 32, 64);          // K 32..63
    v16h bl = load_bt_frag(ws, 64);
    v16h bh = load_bt_frag(ws + 32, 64);
#pragma unroll
    for (int n = 0; n < 4; ++n) {                // pipelined: prefetch n+1 frags
      v16h bln = bl, bhn = bh;
      if (n < 3) {
        bl = load_bt_frag(ws + (n + 1) * 16 * 64, 64);
        bh = load_bt_frag(ws + (n + 1) * 16 * 64 + 32, 64);
      }
      acc[n] = wmma_f16(a0, bln, acc[n]);
      acc[n] = wmma_f16(a1, bhn, acc[n]);
    }
    __syncthreads();
  }

  int rb = row0 + wave * 16 + ((lane >> 4) << 3);
  int cb = lane & 15;
#pragma unroll
  for (int n = 0; n < 4; ++n) {
    int col = col0 + n * 16 + cb;
    float bv = bias[col];
#pragma unroll
    for (int r = 0; r < 8; ++r) {
      int row = rb + r;
      float v = (acc[n][r] + bv) * scale;
      if (mode == 2) {
        out_f[(size_t)row * DEMB + col] = v;
      } else {
        int bidx = row >> 11;
        int t    = row & (TSEQ - 1);
        int h    = col >> 6;
        int hd   = col & (HD - 1);
        size_t bh2 = (size_t)(bidx * NH + h);
        if (mode == 0)
          out_h[(bh2 * TSEQ + t) * HD + hd] = (_Float16)v;
        else                                   // v transposed: [bh][hd][t]
          out_h[(bh2 * HD + hd) * TSEQ + t] = (_Float16)v;
      }
    }
  }
}

// ---------------------------------------------------------------------------
// Flash attention: one block = 8 waves sharing one (b,h); wave owns 16 query
// rows. K chunk [32 x Hd] and V^T chunk [Hd x 32] staged once per block by TDM
// (double buffered). K-frags preloaded before S WMMAs; V-frags issued before
// the softmax VALU block so LDS latency hides under it.
// ---------------------------------------------------------------------------
__global__ void __launch_bounds__(256)
flash_attn(const _Float16* __restrict__ Q, const _Float16* __restrict__ K,
           const _Float16* __restrict__ Vt, const float* __restrict__ mask,
           _Float16* __restrict__ O) {
  __shared__ _Float16 sK[2][32 * HD];    // [key][hd]
  __shared__ _Float16 sV[2][HD * 32];    // [hd][key]  (K-contiguous for B-frags)
  __shared__ _Float16 sP[8][16 * 32];    // per-wave P tile

  int wave = threadIdx.x >> 5;
  int lane = threadIdx.x & 31;
  int bh   = blockIdx.x >> 4;            // 16 blocks per (b,h)
  int qblk = (blockIdx.x & 15) * 8 + wave;
  int b    = bh >> 4;
  int h    = bh & (NH - 1);

  const _Float16* Qp = Q + ((size_t)bh * TSEQ + qblk * 16) * HD;
  const _Float16* Kp = K + (size_t)bh * TSEQ * HD;
  const _Float16* Vp = Vt + (size_t)bh * HD * TSEQ;
  const float*    Mp = mask + ((size_t)b * TSEQ + qblk * 16) * TSEQ;

  v16h aq0 = load_a_frag(Qp, HD);        // Q 16x64 held in registers
  v16h aq1 = load_a_frag(Qp + 32, HD);

  if (wave == 0) {                       // stage chunk 0
    tdm_load_2d(Kp, lds_off(&sK[0][0]), HD, 32, HD, TSEQ, HD);
    tdm_load_2d(Vp, lds_off(&sV[0][0]), 32, HD, TSEQ, HD, TSEQ);
  }

  v8f o0 = {}, o1 = {}, o2 = {}, o3 = {};
  float mrow[8], lrow[8];
#pragma unroll
  for (int r = 0; r < 8; ++r) { mrow[r] = -1e30f; lrow[r] = 0.f; }

  _Float16* pl = &sP[wave][0];
  int rb = (lane >> 4) << 3;
  int cb = lane & 15;
  const int NCH = TSEQ / 32;             // 64 key chunks

  for (int i = 0; i < NCH; ++i) {
    int cur = i & 1, nxt = cur ^ 1;
    if (wave == 0) {
      if (i + 1 < NCH) {
        int kb2 = (i + 1) * 32;
        tdm_load_2d(Kp + (size_t)kb2 * HD, lds_off(&sK[nxt][0]), HD, 32, HD, TSEQ, HD);
        tdm_load_2d(Vp + kb2, lds_off(&sV[nxt][0]), 32, HD, TSEQ, HD, TSEQ);
        __builtin_amdgcn_s_wait_tensorcnt(2);
      } else {
        __builtin_amdgcn_s_wait_tensorcnt(0);
      }
    }
    __syncthreads();

    const _Float16* Kt = &sK[cur][0];
    const _Float16* Vc = &sV[cur][0];
    int kb = i * 32;

    v16h kb0 = load_bt_frag(Kt, HD);                  // keys 0..15, K 0..31
    v16h kb1 = load_bt_frag(Kt + 32, HD);             // keys 0..15, K 32..63
    v16h kb2 = load_bt_frag(Kt + 16 * HD, HD);        // keys 16..31
    v16h kb3 = load_bt_frag(Kt + 16 * HD + 32, HD);

    v8f z = {};
    v8f s0 = wmma_f16(aq0, kb0, z);
    s0     = wmma_f16(aq1, kb1, s0);
    v8f s1 = wmma_f16(aq0, kb2, z);
    s1     = wmma_f16(aq1, kb3, s1);

    // V B-fragments issued early: LDS latency hides under softmax VALU work
    v16h vb0 = load_bt_frag(Vc +  0 * 32, 32);
    v16h vb1 = load_bt_frag(Vc + 16 * 32, 32);
    v16h vb2 = load_bt_frag(Vc + 32 * 32, 32);
    v16h vb3 = load_bt_frag(Vc + 48 * 32, 32);

#pragma unroll
    for (int r = 0; r < 8; ++r) {        // additive mask
      const float* mp = Mp + (size_t)(rb + r) * TSEQ + kb;
      s0[r] += mp[cb];
      s1[r] += mp[16 + cb];
    }

#pragma unroll
    for (int r = 0; r < 8; ++r) {        // online softmax per row
      float mx = fmaxf(s0[r], s1[r]);
#pragma unroll
      for (int off = 8; off; off >>= 1)
        mx = fmaxf(mx, __shfl_xor(mx, off, 32));
      float mn    = fmaxf(mrow[r], mx);
      float alpha = __expf(mrow[r] - mn);
      mrow[r]     = mn;
      float p0 = __expf(s0[r] - mn);
      float p1 = __expf(s1[r] - mn);
      float rs = p0 + p1;
#pragma unroll
      for (int off = 8; off; off >>= 1)
        rs += __shfl_xor(rs, off, 32);
      lrow[r] = lrow[r] * alpha + rs;
      o0[r] *= alpha; o1[r] *= alpha; o2[r] *= alpha; o3[r] *= alpha;
      pl[(rb + r) * 32 + cb]      = (_Float16)p0;
      pl[(rb + r) * 32 + 16 + cb] = (_Float16)p1;
    }

    asm volatile("s_wait_dscnt 0" ::: "memory");   // P RAW in LDS (same wave)
    v16h ap = load_a_frag(pl, 32);

    o0 = wmma_f16(ap, vb0, o0);
    o1 = wmma_f16(ap, vb1, o1);
    o2 = wmma_f16(ap, vb2, o2);
    o3 = wmma_f16(ap, vb3, o3);

    __syncthreads();                     // protect cur buffers before next TDM
  }

  size_t tok0 = (size_t)b * TSEQ + qblk * 16;
#pragma unroll
  for (int r = 0; r < 8; ++r) {
    float inv = 1.f / lrow[r];
    _Float16* op = O + (tok0 + rb + r) * DEMB + h * HD;
    op[cb]      = (_Float16)(o0[r] * inv);
    op[16 + cb] = (_Float16)(o1[r] * inv);
    op[32 + cb] = (_Float16)(o2[r] * inv);
    op[48 + cb] = (_Float16)(o3[r] * inv);
  }
}

// ---------------------------------------------------------------------------
// Launch
// ---------------------------------------------------------------------------
extern "C" void kernel_launch(void* const* d_in, const int* in_sizes, int n_in,
                              void* d_out, int out_size, void* d_ws, size_t ws_size,
                              hipStream_t stream) {
  const float* x    = (const float*)d_in[0];
  const float* mask = (const float*)d_in[1];
  const float* Wq   = (const float*)d_in[2];
  const float* bq   = (const float*)d_in[3];
  const float* Wk   = (const float*)d_in[4];
  const float* bk   = (const float*)d_in[5];
  const float* Wv   = (const float*)d_in[6];
  const float* bv   = (const float*)d_in[7];
  const float* Wo   = (const float*)d_in[8];
  const float* bo   = (const float*)d_in[9];
  float* out = (float*)d_out;

  const size_t dd = (size_t)DEMB * DEMB;
  const size_t td = (size_t)NTOK * DEMB;
  _Float16* Xh  = (_Float16*)d_ws;
  _Float16* Wqt = Xh + td;
  _Float16* Wkt = Wqt + dd;
  _Float16* Wvt = Wkt + dd;
  _Float16* Wot = Wvt + dd;
  _Float16* Qb  = Wot + dd;              // [B*H][T][Hd]
  _Float16* Kb  = Qb + td;               // [B*H][T][Hd]
  _Float16* Vb  = Kb + td;               // [B*H][Hd][T]  (transposed)
  _Float16* Ob  = Vb + td;               // [NTOK][DEMB]

  cvt_f32_to_f16<<<(int)(td / 256), 256, 0, stream>>>(x, Xh, (int)td);

  dim3 tb(64, 4), tg(DEMB / 64, DEMB / 4);
  cvt_transpose_w<<<tg, tb, 0, stream>>>(Wq, Wqt);
  cvt_transpose_w<<<tg, tb, 0, stream>>>(Wk, Wkt);
  cvt_transpose_w<<<tg, tb, 0, stream>>>(Wv, Wvt);
  cvt_transpose_w<<<tg, tb, 0, stream>>>(Wo, Wot);

  dim3 pg(NTOK / 64, DEMB / 64);
  proj_gemm<<<pg, 128, 0, stream>>>(Xh, Wqt, bq, Qb, nullptr, QK_SCALE, 0);
  proj_gemm<<<pg, 128, 0, stream>>>(Xh, Wkt, bk, Kb, nullptr, 1.0f, 0);
  proj_gemm<<<pg, 128, 0, stream>>>(Xh, Wvt, bv, Vb, nullptr, 1.0f, 1);

  flash_attn<<<NB * NH * 16, 256, 0, stream>>>(Qb, Kb, Vb, mask, Ob);

  proj_gemm<<<pg, 128, 0, stream>>>(Ob, Wot, bo, nullptr, out, 1.0f, 2);
}
